// SingleHeadAttention_31868657336870
// MI455X (gfx1250) — matrix operand order repaired and checked by reading, not compiled, not run on hardware
//
#include <hip/hip_runtime.h>

#define B_  8
#define T_  2048
#define D_  768
#define HS_ 128

typedef __bf16 bf16_t;
typedef __attribute__((ext_vector_type(16))) __bf16 v16bf;
typedef __attribute__((ext_vector_type(8)))  float  v8f;
typedef __attribute__((ext_vector_type(4)))  unsigned int v4u;
typedef __attribute__((ext_vector_type(8)))  int v8i;
typedef __attribute__((ext_vector_type(4)))  int v4i;

union Frag {
  v16bf  v;
  uint4  q[2];
  bf16_t b[16];
};

union Pack8 {
  uint4  q;
  bf16_t b[8];
};

__device__ __forceinline__ v8f wmma_bf16(v16bf a, v16bf b, v8f c) {
  // D = A(16x32 bf16) * B(32x16 bf16) + C(16x16 f32)
  return __builtin_amdgcn_wmma_f32_16x16x32_bf16(false, a, false, b, (short)0, c,
                                                 false, false);
}

// ---------------------------------------------------------------------------
// TDM: 2D bf16 tile Global -> LDS  (D# per CDNA5 ISA ch.8; data_size=2B)
// dw0flags carries data_size + LDS pad fields (pad staggers LDS banks).
// clang-23 toolchain: 6-arg builtin (extra int32x8 group, zero-filled).
// ---------------------------------------------------------------------------
__device__ __forceinline__ void tdm_load_2d(unsigned lds_addr, const void* gptr,
                                            unsigned tensor_d0, unsigned tensor_d1,
                                            unsigned tile_d0, unsigned tile_d1,
                                            unsigned stride0, unsigned dw0flags) {
  unsigned long long ga = (unsigned long long)(size_t)gptr;
  // group0: [1:0]=count=1 | [63:32]=lds_addr | [120:64]=global_addr | [127:126]=2
  v4u g0 = {1u, lds_addr, (unsigned)ga,
            (unsigned)((ga >> 32) & 0x1FFFFFFu) | 0x80000000u};
  // group1 bitfields (see ISA 8.4)
  v8i g1;
  g1[0] = (int)dw0flags;                               // mask=0,data_size,pads
  g1[1] = (int)((tensor_d0 & 0xFFFFu) << 16);          // atomic_addr=0|dim0 lo
  g1[2] = (int)((tensor_d0 >> 16) | ((tensor_d1 & 0xFFFFu) << 16));
  g1[3] = (int)((tensor_d1 >> 16) | (tile_d0 << 16));  // dim1 hi | tile_dim0
  g1[4] = (int)(tile_d1 & 0xFFFFu);                    // tile_dim1 | tile_dim2=0
  g1[5] = (int)stride0;                                // dim0 stride lo
  g1[6] = 0;                                           // stride hi | stride1
  g1[7] = 0;
  v4i z4 = {0, 0, 0, 0};
  v8i z8 = {0, 0, 0, 0, 0, 0, 0, 0};
  __builtin_amdgcn_tensor_load_to_lds(g0, g1, z4, z4, z8, 0);
}

// dw0: data_size=1(2B) at [17:16]; pad_en bit20; pad_interval [24:22]; amount [31:25]
#define TDM_FLAGS_K 0x07510000u  // pad every 64 dwords (256B row) by 4 dwords
#define TDM_FLAGS_V 0x07110000u  // pad every 32 dwords (128B row) by 4 dwords

// ---------------------------------------------------------------------------
// Kernel 0: weights -> bf16 W^T [3][128][768], biases -> f32 [3][128]
// ---------------------------------------------------------------------------
__global__ void __launch_bounds__(256) prep_weights(
    const float* __restrict__ Wq, const float* __restrict__ bq,
    const float* __restrict__ Wk, const float* __restrict__ bk,
    const float* __restrict__ Wv, const float* __restrict__ bv,
    bf16_t* __restrict__ Wt, float* __restrict__ bias) {
  int idx = blockIdx.x * blockDim.x + threadIdx.x;
  if (idx < 3 * HS_ * D_) {
    int m   = idx / (HS_ * D_);
    int rem = idx % (HS_ * D_);
    int c   = rem / D_;
    int kk  = rem % D_;
    const float* W = (m == 0) ? Wq : (m == 1) ? Wk : Wv;
    Wt[idx] = (bf16_t)W[kk * HS_ + c];   // transpose: Wt[m][c][kk] = W[kk][c]
  }
  if (idx < 3 * HS_) {
    const float* bsrc = (idx < HS_) ? bq : (idx < 2 * HS_) ? bk : bv;
    bias[idx] = bsrc[idx % HS_];
  }
}

// ---------------------------------------------------------------------------
// Kernel 1: q/k/v projections via WMMA.  Block = 8 waves; grid = rowTiles*3.
// Wave computes one 16(token) x 16(col) tile of one of {q,k,vT}, K=768.
// q,k -> [B*T][128] bf16 row-major ;  v -> vT [B][128][T] bf16.
// ---------------------------------------------------------------------------
__global__ void __launch_bounds__(256) qkv_proj(
    const float*  __restrict__ x,
    const bf16_t* __restrict__ Wt,    // [3][128][768]
    const float*  __restrict__ bias,  // [3][128]
    bf16_t* __restrict__ qo,
    bf16_t* __restrict__ ko,
    bf16_t* __restrict__ vT) {
  const int tid  = threadIdx.x;
  const int wv   = tid >> 5;
  const int ln   = tid & 15;
  const int h    = (tid >> 4) & 1;
  const int rt   = blockIdx.x / 3;
  const int part = blockIdx.x % 3;
  const int t24  = part * 8 + wv;      // 0..23
  const int m    = t24 >> 3;           // matrix: 0=q 1=k 2=v
  const int ct   = t24 & 7;            // col tile (hs/16)

  const float*  xa = x + (size_t)(rt * 16 + ln) * D_;             // A row/lane
  const bf16_t* bt = Wt + ((size_t)m * HS_ + ct * 16 + ln) * D_;  // B^T row

  v8f acc = {0.f, 0.f, 0.f, 0.f, 0.f, 0.f, 0.f, 0.f};
#pragma unroll 4
  for (int kc = 0; kc < D_ / 32; ++kc) {
    Frag A, Bf;
    const float* pa = xa + kc * 32 + h * 8;
#pragma unroll
    for (int i = 0; i < 8; ++i) A.b[i]     = (bf16_t)pa[i];       // K = h*8..
#pragma unroll
    for (int i = 0; i < 8; ++i) A.b[8 + i] = (bf16_t)pa[16 + i];  // K = 16+h*8
    const uint4* pb = (const uint4*)(bt + kc * 32 + h * 8);
    Bf.q[0] = pb[0];
    Bf.q[1] = pb[2];
    acc = wmma_bf16(A.v, Bf.v, acc);
  }
  const float bcol = bias[m * HS_ + ct * 16 + ln];
#pragma unroll
  for (int r = 0; r < 8; ++r) acc[r] += bcol;

  if (m < 2) {
    bf16_t* dst = (m == 0) ? qo : ko;
#pragma unroll
    for (int r = 0; r < 8; ++r) {
      int tok = rt * 16 + 8 * h + r;     // D layout: row = r + 8*h
      dst[(size_t)tok * HS_ + ct * 16 + ln] = (bf16_t)acc[r];
    }
  } else {
    int tok0 = rt * 16 + 8 * h;          // 8 consecutive tokens -> one b128
    int bb   = tok0 / T_;
    int t0   = tok0 - bb * T_;
    int c    = ct * 16 + ln;
    Pack8 pk;
#pragma unroll
    for (int r = 0; r < 8; ++r) pk.b[r] = (bf16_t)acc[r];
    *(uint4*)(vT + ((size_t)bb * HS_ + c) * T_ + t0) = pk.q;
  }
}

// ---------------------------------------------------------------------------
// Kernel 2: causal flash attention with TDM-staged K/V in LDS.
// Grid = B * T/128, block = 8 waves (wave owns 16 query rows).
// Wave 0 DMA's the next 64-key K tile and V^T tile into the alternate LDS
// buffer (tensor_load_to_lds, TENSORcnt) while all waves compute on the
// current one; sync = s_wait_tensorcnt(0) + block barrier.
// ---------------------------------------------------------------------------
__global__ void __launch_bounds__(256) flash_attn(
    const bf16_t* __restrict__ qg,
    const bf16_t* __restrict__ kg,
    const bf16_t* __restrict__ vTg,
    float* __restrict__ out) {
  // TDM pads each K row (256B) / V row (128B) by 16B -> bank-staggered rows
  __shared__ __align__(16) bf16_t Kbuf[2][64][136];   // 34.0 KB
  __shared__ __align__(16) bf16_t Vbuf[2][128][72];   // 36.0 KB
  __shared__ __align__(16) bf16_t pbuf[8][16][72];    // 18.0 KB (per-wave P)

  const int tid = threadIdx.x;
  const int wv  = tid >> 5;
  const int ln  = tid & 15;
  const int h   = (tid >> 4) & 1;
  const int b   = blockIdx.x >> 4;     // T/128 = 16 q-tiles per batch
  const int qt  = blockIdx.x & 15;
  const int qs  = qt * 128 + wv * 16;  // wave's first query row

  const bf16_t* qp = qg  + ((size_t)b * T_ + qs) * HS_;
  const bf16_t* kp = kg  + (size_t)b * T_ * HS_;
  const bf16_t* vT = vTg + (size_t)b * HS_ * T_;

  // Q fragments: 16 x 128 bf16 = 4 A-frags of K=32
  Frag Qf[4];
#pragma unroll
  for (int kc = 0; kc < 4; ++kc) {
    const bf16_t* pr = qp + (size_t)ln * HS_ + kc * 32 + h * 8;
    Qf[kc].q[0] = *(const uint4*)pr;
    Qf[kc].q[1] = *(const uint4*)(pr + 16);
  }

  const v8f vzero = {0.f, 0.f, 0.f, 0.f, 0.f, 0.f, 0.f, 0.f};
  v8f O[8];
#pragma unroll
  for (int nt = 0; nt < 8; ++nt) O[nt] = vzero;

  const float NEG_INF = -__builtin_inff();
  float mrow[8], lrow[8];
#pragma unroll
  for (int r = 0; r < 8; ++r) { mrow[r] = NEG_INF; lrow[r] = 0.f; }

  const float kScale = 0.1275253083f;  // log2(e)/sqrt(128): exp2-domain
  const int lastkt = (qt * 128 + 127) >> 6;  // uniform over block: 2*qt+1

  // --- prologue: stage tile 0 ---
  if (wv == 0) {
    tdm_load_2d((unsigned)(size_t)&Kbuf[0][0][0], kp, HS_, T_, HS_, 64, HS_,
                TDM_FLAGS_K);
    tdm_load_2d((unsigned)(size_t)&Vbuf[0][0][0], vT, T_, HS_, 64, HS_, T_,
                TDM_FLAGS_V);
    __builtin_amdgcn_s_wait_tensorcnt(0);
  }
  __syncthreads();

  for (int kt = 0; kt <= lastkt; ++kt) {
    const int kbase = kt << 6;
    const int cur   = kt & 1;

    // --- async: stage tile kt+1 into the other buffer ---
    if (wv == 0 && kt < lastkt) {
      const int nb = kbase + 64;
      tdm_load_2d((unsigned)(size_t)&Kbuf[cur ^ 1][0][0],
                  kp + (size_t)nb * HS_, HS_, T_, HS_, 64, HS_, TDM_FLAGS_K);
      tdm_load_2d((unsigned)(size_t)&Vbuf[cur ^ 1][0][0],
                  vT + nb, T_, HS_, 64, HS_, T_, TDM_FLAGS_V);
    }

    if (qs + 15 >= kbase) {  // wave's causal extent covers this tile
      // ---- S = Q K^T : 4 col-subtiles x 4 K-steps = 16 WMMA ----
      v8f S[4];
#pragma unroll
      for (int sub = 0; sub < 4; ++sub) {
        S[sub] = vzero;
#pragma unroll
        for (int kc = 0; kc < 4; ++kc) {
          Frag Kf;  // B-frag: column = key, B^T row = K row (from LDS)
          const bf16_t* kr = &Kbuf[cur][sub * 16 + ln][kc * 32 + h * 8];
          Kf.q[0] = ((const uint4*)kr)[0];
          Kf.q[1] = ((const uint4*)kr)[2];
          S[sub] = wmma_bf16(Qf[kc].v, Kf.v, S[sub]);
        }
      }

      // ---- causal mask + scale (exp2 domain) + row max ----
      const bool needmask = (kbase + 63) > qs;
      float tmax[8];
#pragma unroll
      for (int r = 0; r < 8; ++r) tmax[r] = NEG_INF;
#pragma unroll
      for (int sub = 0; sub < 4; ++sub) {
        const int key = kbase + sub * 16 + ln;
#pragma unroll
        for (int r = 0; r < 8; ++r) {     // D layout: row = r + 8*h
          float s = S[sub][r] * kScale;
          if (needmask && key > qs + 8 * h + r) s = NEG_INF;
          S[sub][r] = s;
          tmax[r]   = fmaxf(tmax[r], s);
        }
      }
#pragma unroll
      for (int off = 1; off < 16; off <<= 1)
#pragma unroll
        for (int r = 0; r < 8; ++r)
          tmax[r] = fmaxf(tmax[r], __shfl_xor(tmax[r], off, 32));

      // ---- online softmax rescale ----
      float alpha[8];
#pragma unroll
      for (int r = 0; r < 8; ++r) {
        float mo = mrow[r];
        float mn = fmaxf(mo, tmax[r]);
        mrow[r]  = mn;
        alpha[r] = exp2f(mo - mn);
        lrow[r] *= alpha[r];
      }
#pragma unroll
      for (int nt = 0; nt < 8; ++nt)
#pragma unroll
        for (int r = 0; r < 8; ++r) O[nt][r] *= alpha[r];

      // ---- P = exp2(S - m), row sums, spill to LDS for layout change ----
      float rsum[8];
#pragma unroll
      for (int r = 0; r < 8; ++r) rsum[r] = 0.f;
#pragma unroll
      for (int sub = 0; sub < 4; ++sub)
#pragma unroll
        for (int r = 0; r < 8; ++r) {
          float p = exp2f(S[sub][r] - mrow[r]);
          rsum[r] += p;
          pbuf[wv][8 * h + r][sub * 16 + ln] = (bf16_t)p;
        }
#pragma unroll
      for (int off = 1; off < 16; off <<= 1)
#pragma unroll
        for (int r = 0; r < 8; ++r) rsum[r] += __shfl_xor(rsum[r], off, 32);
#pragma unroll
      for (int r = 0; r < 8; ++r) lrow[r] += rsum[r];

      // ---- reload P as A-fragments (per-wave slab, DS in-order) ----
      Frag Pf[2];
#pragma unroll
      for (int pc = 0; pc < 2; ++pc) {
        const bf16_t* pr = &pbuf[wv][ln][pc * 32 + h * 8];
        Pf[pc].q[0] = *(const uint4*)pr;
        Pf[pc].q[1] = *(const uint4*)(pr + 16);
      }

      // ---- O += P V : 8 hs-subtiles x 2 K-steps = 16 WMMA ----
#pragma unroll
      for (int nt = 0; nt < 8; ++nt) {
#pragma unroll
        for (int pc = 0; pc < 2; ++pc) {
          Frag Vf;  // B-frag: column = hs dim, B^T row = vT row (from LDS)
          const bf16_t* vr = &Vbuf[cur][nt * 16 + ln][pc * 32 + h * 8];
          Vf.q[0] = ((const uint4*)vr)[0];
          Vf.q[1] = ((const uint4*)vr)[2];
          O[nt] = wmma_bf16(Pf[pc].v, Vf.v, O[nt]);
        }
      }
    }

    // --- drain the async stage for tile kt+1, then block-wide hand-off ---
    if (wv == 0 && kt < lastkt) __builtin_amdgcn_s_wait_tensorcnt(0);
    __syncthreads();
  }

  // ---- epilogue: O / l -> out f32 ----
  float inv[8];
#pragma unroll
  for (int r = 0; r < 8; ++r) inv[r] = 1.0f / lrow[r];
#pragma unroll
  for (int nt = 0; nt < 8; ++nt)
#pragma unroll
    for (int r = 0; r < 8; ++r)
      out[((size_t)b * T_ + qs + 8 * h + r) * HS_ + nt * 16 + ln] =
          O[nt][r] * inv[r];
}

// ---------------------------------------------------------------------------
// Workspace layout (bytes):
//   0        : Wt  bf16 [3][128][768]   = 589824
//   589824   : bias f32 [3][128]        = 1536
//   591872   : q   bf16 [B*T][128]      = 4194304
//   4786176  : k   bf16 [B*T][128]      = 4194304
//   8980480  : vT  bf16 [B][128][T]     = 4194304
//   total ~ 12.6 MB
// ---------------------------------------------------------------------------
extern "C" void kernel_launch(void* const* d_in, const int* in_sizes, int n_in,
                              void* d_out, int out_size, void* d_ws,
                              size_t ws_size, hipStream_t stream) {
  const float* x  = (const float*)d_in[0];
  const float* Wq = (const float*)d_in[1];
  const float* bq = (const float*)d_in[2];
  const float* Wk = (const float*)d_in[3];
  const float* bk = (const float*)d_in[4];
  const float* Wv = (const float*)d_in[5];
  const float* bv = (const float*)d_in[6];

  char* ws = (char*)d_ws;
  bf16_t* Wt   = (bf16_t*)(ws + 0);
  float*  bias = (float*)(ws + 589824);
  bf16_t* q    = (bf16_t*)(ws + 591872);
  bf16_t* k    = (bf16_t*)(ws + 591872 + 4194304);
  bf16_t* vT   = (bf16_t*)(ws + 591872 + 2 * 4194304);
  float*  out  = (float*)d_out;

  prep_weights<<<(3 * HS_ * D_ + 255) / 256, 256, 0, stream>>>(Wq, bq, Wk, bk,
                                                               Wv, bv, Wt, bias);
  qkv_proj<<<(B_ * T_ / 16) * 3, 256, 0, stream>>>(x, Wt, bias, q, k, vT);
  flash_attn<<<B_ * (T_ / 128), 256, 0, stream>>>(q, k, vT, out);
}
